// Model_3796751090165
// MI455X (gfx1250) — compile-verified, matrix operand order checked by reading, not run on hardware
//
#include <hip/hip_runtime.h>
#include <hip/hip_bf16.h>

// ---------------------------------------------------------------------------
// CDNA5 WMMA types (wave32): D(16x16 f32) = A(16x32 bf16) x B(32x16 bf16) + C
// ---------------------------------------------------------------------------
typedef __attribute__((ext_vector_type(16))) __bf16 bf16x16;
typedef __attribute__((ext_vector_type(8)))  __bf16 bf16x8;
typedef __attribute__((ext_vector_type(8)))  float  f32x8;

#define LN_EPS 1e-5f

__device__ __forceinline__ bf16x16 cat8(bf16x8 lo, bf16x8 hi)
{
    return __builtin_shufflevector(lo, hi,
        0, 1, 2, 3, 4, 5, 6, 7, 8, 9, 10, 11, 12, 13, 14, 15);
}

// ---------------------------------------------------------------------------
// Generic dense GEMM on pre-packed bf16:  C[M,N] = act( A[M,K] @ B[N,K]^T + b )
// K is the fast axis of both operands, so each WMMA fragment is two 16-byte
// loads per lane (CDNA5 16-bit A layout: lane group g holds K=g*8..g*8+7 in
// elements 0..7 and K=16+g*8..+7 in elements 8..15).  One wave computes a
// 16 x (16*NT) tile; the A fragment is reused across NT WMMAs per K-step.
// M%16==0, N%(16*NT)==0, K%32==0 for all call sites -> no guards.
// ---------------------------------------------------------------------------
template <int NT, bool RELU, bool HAS_BIAS>
__global__ void gemm_bf16_wmma(const __bf16* __restrict__ A,   // [M,K]
                               const __bf16* __restrict__ B,   // [N,K]
                               const float* __restrict__ bias, // [N] or null
                               float* __restrict__ C,          // [M,N]
                               int Mdim, int Ndim, int Kdim)
{
    const int lane  = threadIdx.x;
    const int tileM = blockIdx.y;
    const int tileN = blockIdx.x;
    const int g     = lane >> 4;
    const int mn    = lane & 15;

    const __bf16* Arow = A + (size_t)(tileM * 16 + mn) * Kdim;

    const __bf16* Brow[NT];
#pragma unroll
    for (int t = 0; t < NT; ++t)
        Brow[t] = B + (size_t)((tileN * NT + t) * 16 + mn) * Kdim;

    f32x8 acc[NT];
#pragma unroll
    for (int t = 0; t < NT; ++t) acc[t] = (f32x8){};

    for (int k0 = 0; k0 < Kdim; k0 += 32) {
        if (k0 + 32 < Kdim)                         // global_prefetch_b8
            __builtin_prefetch(Arow + k0 + 32, 0, 1);

        const bf16x16 a = cat8(*(const bf16x8*)(Arow + k0 + g * 8),
                               *(const bf16x8*)(Arow + k0 + 16 + g * 8));
#pragma unroll
        for (int t = 0; t < NT; ++t) {
            const bf16x16 b = cat8(*(const bf16x8*)(Brow[t] + k0 + g * 8),
                                   *(const bf16x8*)(Brow[t] + k0 + 16 + g * 8));
            acc[t] = __builtin_amdgcn_wmma_f32_16x16x32_bf16(
                false, a, false, b, (short)0, acc[t], false, false);
        }
    }

#pragma unroll
    for (int t = 0; t < NT; ++t) {
        const int ncol = (tileN * NT + t) * 16 + mn;
        const float bv = HAS_BIAS ? bias[ncol] : 0.0f;
#pragma unroll
        for (int r = 0; r < 8; ++r) {
            const int mout = tileM * 16 + g * 8 + r;
            float v = acc[t][r] + bv;
            if (RELU) v = fmaxf(v, 0.0f);
            C[(size_t)mout * Ndim + ncol] = v;
        }
    }
}

// ---------------------------------------------------------------------------
// Final GEMM:  C[M,N] = A[M,KD] @ B[N,KD]^T  with LDS-staged B stripe.
// Block = NWAVES waves (NWAVES*32 threads) covering (NWAVES*16) x (NT*16).
// The B stripe (NT*16 rows x KD) is loaded to LDS once (KD=128 -> 20KB),
// then all waves read fragments via ds_load_b128, cutting L2 read traffic
// by NWAVES x on the dominant 10000x10000x128 product (HBM floor = 400MB
// output store @ 23.3 TB/s).
// ---------------------------------------------------------------------------
template <int NWAVES, int NT, int KD>
__global__ void gemm_final_wmma(const __bf16* __restrict__ A,   // [M,KD]
                                const __bf16* __restrict__ B,   // [N,KD]
                                float* __restrict__ C, int Ndim)
{
    constexpr int BN = NT * 16;                    // cols per block
    __shared__ __bf16 sB[BN * KD];                 // 80*128*2 = 20 KB

    const int tid  = threadIdx.x;                  // 0 .. NWAVES*32-1
    const int wv   = tid >> 5;
    const int lane = tid & 31;
    const int g    = lane >> 4;
    const int mn   = lane & 15;

    const int nBase = blockIdx.x * BN;
    const int mBase = blockIdx.y * (NWAVES * 16);

    // ---- cooperative stage of B stripe into LDS (16B chunks) ----
    {
        const __bf16* src = B + (size_t)nBase * KD;
        constexpr int nChunks = BN * KD / 8;       // 1280
#pragma unroll
        for (int c = tid; c < nChunks; c += NWAVES * 32)
            *(bf16x8*)(sB + c * 8) = *(const bf16x8*)(src + (size_t)c * 8);
    }
    __syncthreads();

    const __bf16* Arow = A + (size_t)(mBase + wv * 16 + mn) * KD;

    f32x8 acc[NT];
#pragma unroll
    for (int t = 0; t < NT; ++t) acc[t] = (f32x8){};

#pragma unroll
    for (int k0 = 0; k0 < KD; k0 += 32) {
        const bf16x16 a = cat8(*(const bf16x8*)(Arow + k0 + g * 8),
                               *(const bf16x8*)(Arow + k0 + 16 + g * 8));
#pragma unroll
        for (int t = 0; t < NT; ++t) {
            const __bf16* brow = sB + (t * 16 + mn) * KD;
            const bf16x16 b = cat8(*(const bf16x8*)(brow + k0 + g * 8),
                                   *(const bf16x8*)(brow + k0 + 16 + g * 8));
            acc[t] = __builtin_amdgcn_wmma_f32_16x16x32_bf16(
                false, a, false, b, (short)0, acc[t], false, false);
        }
    }

#pragma unroll
    for (int t = 0; t < NT; ++t) {
        const int ncol = nBase + t * 16 + mn;
#pragma unroll
        for (int r = 0; r < 8; ++r) {
            const int mout = mBase + wv * 16 + g * 8 + r;
            C[(size_t)mout * Ndim + ncol] = acc[t][r];
        }
    }
}

// ---------------------------------------------------------------------------
// Packing helpers (run once per operand; tiny vs GEMM cost)
// ---------------------------------------------------------------------------
__global__ void pack_bf16(const float* __restrict__ src, __bf16* __restrict__ dst,
                          size_t n)
{
    size_t i = (size_t)blockIdx.x * blockDim.x + threadIdx.x;
    if (i < n) dst[i] = (__bf16)src[i];
}

// src [R,C] fp32 -> dst [C,R] bf16  (GCN weights [in,out] -> [N,K])
__global__ void transpose_pack_bf16(const float* __restrict__ src,
                                    __bf16* __restrict__ dst, int R, int Cn)
{
    int i = blockIdx.x * blockDim.x + threadIdx.x;
    if (i < R * Cn) {
        const int r = i / Cn, c = i % Cn;
        dst[(size_t)c * R + r] = (__bf16)src[i];
    }
}

// ---------------------------------------------------------------------------
// Graph helpers
// ---------------------------------------------------------------------------
__global__ void zero_f32(float* __restrict__ p, size_t n)
{
    size_t i = (size_t)blockIdx.x * blockDim.x + threadIdx.x;
    if (i < n) p[i] = 0.0f;
}

__global__ void deg_accum(const int* __restrict__ col, const float* __restrict__ w,
                          float* __restrict__ deg, int E)
{
    int e = blockIdx.x * blockDim.x + threadIdx.x;
    if (e < E) atomicAdd(&deg[col[e]], w[e]);
}

__global__ void rsqrt_deg(float* __restrict__ d, int n)
{
    int i = blockIdx.x * blockDim.x + threadIdx.x;
    if (i < n) d[i] = rsqrtf(d[i] + 1.0f);    // +1 self loop
}

__global__ void scatter_edges(const int* __restrict__ row, const int* __restrict__ col,
                              const float* __restrict__ w, const float* __restrict__ dis,
                              const float* __restrict__ h, float* __restrict__ agg,
                              int F)
{
    const int e = blockIdx.x;
    const int r = row[e];
    const int c = col[e];
    const float norm = dis[r] * w[e] * dis[c];
    const int f = threadIdx.x;
    atomicAdd(&agg[(size_t)c * F + f], norm * h[(size_t)r * F + f]);
}

// v = agg + d_is^2*h + bias ; ReLU ; LayerNorm -> out (block per node, LDS tree)
__global__ void gcn_epilogue_ln(const float* __restrict__ agg, const float* __restrict__ h,
                                const float* __restrict__ dis, const float* __restrict__ bias,
                                const float* __restrict__ gamma, const float* __restrict__ beta,
                                float* __restrict__ out, int F)
{
    extern __shared__ float red[];
    const int n = blockIdx.x;
    const int f = threadIdx.x;

    float sn = dis[n];
    sn *= sn;
    float v = agg[(size_t)n * F + f] + sn * h[(size_t)n * F + f] + bias[f];
    v = fmaxf(v, 0.0f);

    red[f] = v;
    __syncthreads();
    for (int s = F >> 1; s > 0; s >>= 1) {
        if (f < s) red[f] += red[f + s];
        __syncthreads();
    }
    const float mu = red[0] / (float)F;
    __syncthreads();

    const float d = v - mu;
    red[f] = d * d;
    __syncthreads();
    for (int s = F >> 1; s > 0; s >>= 1) {
        if (f < s) red[f] += red[f + s];
        __syncthreads();
    }
    const float var = red[0] / (float)F;

    out[(size_t)n * F + f] = d * rsqrtf(var + LN_EPS) * gamma[f] + beta[f];
}

// ---------------------------------------------------------------------------
// One branch: 2x(GCN -> ReLU -> LN) then ReLU MLP 256->256->128->128
// ---------------------------------------------------------------------------
static void run_branch(const float* x_in, const int* ei, const float* ew,
                       const float* Wg1, const float* bg1,
                       const float* Wg2, const float* bg2,
                       const float* ln1g, const float* ln1b,
                       const float* ln2g, const float* ln2b,
                       const float* Wl1, const float* bl1,
                       const float* Wl2, const float* bl2,
                       const float* Wl3, const float* bl3,
                       float* dis, float* h1, float* agg1, float* x1,
                       float* h2, float* agg2, float* xln,
                       float* t1, float* t2,
                       __bf16* Ab, __bf16* Bb, __bf16* featb,
                       hipStream_t stream)
{
    const int N = 10000, E = 320000;
    const int* row = ei;
    const int* col = ei + E;
    const size_t NB = 255;

    // symmetric normalization: d_is = rsqrt(sum_w + 1)
    zero_f32<<<(N + 255) / 256, 256, 0, stream>>>(dis, (size_t)N);
    deg_accum<<<(E + 255) / 256, 256, 0, stream>>>(col, ew, dis, E);
    rsqrt_deg<<<(N + 255) / 256, 256, 0, stream>>>(dis, N);

    // --- GCN 1: h1 = x @ Wg1   (256 -> 512) --------------------------------
    pack_bf16<<<((size_t)N * 256 + NB) / 256, 256, 0, stream>>>(x_in, Ab, (size_t)N * 256);
    transpose_pack_bf16<<<(256 * 512 + 255) / 256, 256, 0, stream>>>(Wg1, Bb, 256, 512);
    gemm_bf16_wmma<4, false, false>
        <<<dim3(512 / 64, N / 16), 32, 0, stream>>>(Ab, Bb, nullptr, h1, N, 512, 256);
    zero_f32<<<((size_t)N * 512 + NB) / 256, 256, 0, stream>>>(agg1, (size_t)N * 512);
    scatter_edges<<<E, 512, 0, stream>>>(row, col, ew, dis, h1, agg1, 512);
    gcn_epilogue_ln<<<N, 512, 512 * sizeof(float), stream>>>(
        agg1, h1, dis, bg1, ln1g, ln1b, x1, 512);

    // --- GCN 2: h2 = x1 @ Wg2  (512 -> 256) --------------------------------
    pack_bf16<<<((size_t)N * 512 + NB) / 256, 256, 0, stream>>>(x1, Ab, (size_t)N * 512);
    transpose_pack_bf16<<<(512 * 256 + 255) / 256, 256, 0, stream>>>(Wg2, Bb, 512, 256);
    gemm_bf16_wmma<4, false, false>
        <<<dim3(256 / 64, N / 16), 32, 0, stream>>>(Ab, Bb, nullptr, h2, N, 256, 512);
    zero_f32<<<((size_t)N * 256 + NB) / 256, 256, 0, stream>>>(agg2, (size_t)N * 256);
    scatter_edges<<<E, 256, 0, stream>>>(row, col, ew, dis, h2, agg2, 256);
    gcn_epilogue_ln<<<N, 256, 256 * sizeof(float), stream>>>(
        agg2, h2, dis, bg2, ln2g, ln2b, xln, 256);

    // --- MLP head (torch Linear weights already [N,K]) ----------------------
    pack_bf16<<<((size_t)N * 256 + NB) / 256, 256, 0, stream>>>(xln, Ab, (size_t)N * 256);
    pack_bf16<<<(256 * 256 + 255) / 256, 256, 0, stream>>>(Wl1, Bb, (size_t)256 * 256);
    gemm_bf16_wmma<4, true, true>
        <<<dim3(256 / 64, N / 16), 32, 0, stream>>>(Ab, Bb, bl1, t1, N, 256, 256);

    pack_bf16<<<((size_t)N * 256 + NB) / 256, 256, 0, stream>>>(t1, Ab, (size_t)N * 256);
    pack_bf16<<<(128 * 256 + 255) / 256, 256, 0, stream>>>(Wl2, Bb, (size_t)128 * 256);
    gemm_bf16_wmma<4, true, true>
        <<<dim3(128 / 64, N / 16), 32, 0, stream>>>(Ab, Bb, bl2, t2, N, 128, 256);

    pack_bf16<<<((size_t)N * 128 + NB) / 256, 256, 0, stream>>>(t2, Ab, (size_t)N * 128);
    pack_bf16<<<(128 * 128 + 255) / 256, 256, 0, stream>>>(Wl3, Bb, (size_t)128 * 128);
    gemm_bf16_wmma<4, true, true>
        <<<dim3(128 / 64, N / 16), 32, 0, stream>>>(Ab, Bb, bl3, t1, N, 128, 128);

    // pack final features to bf16 for the big X @ Y^T GEMM
    pack_bf16<<<((size_t)N * 128 + NB) / 256, 256, 0, stream>>>(t1, featb, (size_t)N * 128);
}

// ---------------------------------------------------------------------------
// Entry point
// ---------------------------------------------------------------------------
extern "C" void kernel_launch(void* const* d_in, const int* in_sizes, int n_in,
                              void* d_out, int out_size, void* d_ws, size_t ws_size,
                              hipStream_t stream)
{
    (void)in_sizes; (void)n_in; (void)out_size; (void)ws_size;
    const int N = 10000;

    const float* x_m  = (const float*)d_in[0];
    const float* x_d  = (const float*)d_in[1];
    const int*   mmEI = (const int*)  d_in[2];
    const float* mmEW = (const float*)d_in[3];
    const int*   ddEI = (const int*)  d_in[4];
    const float* ddEW = (const float*)d_in[5];
    const float* Wg1  = (const float*)d_in[6];  const float* bg1 = (const float*)d_in[7];
    const float* Wg2  = (const float*)d_in[8];  const float* bg2 = (const float*)d_in[9];
    const float* Wd1  = (const float*)d_in[10]; const float* bd1 = (const float*)d_in[11];
    const float* Wd2  = (const float*)d_in[12]; const float* bd2 = (const float*)d_in[13];
    const float* lnx1g = (const float*)d_in[14]; const float* lnx1b = (const float*)d_in[15];
    const float* lnx2g = (const float*)d_in[16]; const float* lnx2b = (const float*)d_in[17];
    const float* lny1g = (const float*)d_in[18]; const float* lny1b = (const float*)d_in[19];
    const float* lny2g = (const float*)d_in[20]; const float* lny2b = (const float*)d_in[21];
    const float* Wlx1 = (const float*)d_in[22]; const float* blx1 = (const float*)d_in[23];
    const float* Wlx2 = (const float*)d_in[24]; const float* blx2 = (const float*)d_in[25];
    const float* Wlx3 = (const float*)d_in[26]; const float* blx3 = (const float*)d_in[27];
    const float* Wly1 = (const float*)d_in[28]; const float* bly1 = (const float*)d_in[29];
    const float* Wly2 = (const float*)d_in[30]; const float* bly2 = (const float*)d_in[31];
    const float* Wly3 = (const float*)d_in[32]; const float* bly3 = (const float*)d_in[33];

    // ---- workspace carve-up (bytes, 1KB aligned) ----
    char* wsb = (char*)d_ws;
    size_t off = 0;
    auto carve = [&](size_t bytes) {
        void* p = wsb + off;
        off += (bytes + 1023) & ~(size_t)1023;
        return p;
    };
    float*  dis   = (float*)carve((size_t)N * sizeof(float));
    float*  h1    = (float*)carve((size_t)N * 512 * sizeof(float));
    float*  agg1  = (float*)carve((size_t)N * 512 * sizeof(float));
    float*  x1    = (float*)carve((size_t)N * 512 * sizeof(float));
    float*  h2    = (float*)carve((size_t)N * 256 * sizeof(float));
    float*  agg2  = (float*)carve((size_t)N * 256 * sizeof(float));
    float*  xln   = (float*)carve((size_t)N * 256 * sizeof(float));
    float*  t1    = (float*)carve((size_t)N * 256 * sizeof(float));
    float*  t2    = (float*)carve((size_t)N * 128 * sizeof(float));
    __bf16* Ab    = (__bf16*)carve((size_t)N * 512 * sizeof(__bf16));
    __bf16* Bb    = (__bf16*)carve((size_t)512 * 512 * sizeof(__bf16));
    __bf16* Xfb   = (__bf16*)carve((size_t)N * 128 * sizeof(__bf16));
    __bf16* Yfb   = (__bf16*)carve((size_t)N * 128 * sizeof(__bf16));

    run_branch(x_m, mmEI, mmEW, Wg1, bg1, Wg2, bg2,
               lnx1g, lnx1b, lnx2g, lnx2b,
               Wlx1, blx1, Wlx2, blx2, Wlx3, blx3,
               dis, h1, agg1, x1, h2, agg2, xln, t1, t2,
               Ab, Bb, Xfb, stream);

    run_branch(x_d, ddEI, ddEW, Wd1, bd1, Wd2, bd2,
               lny1g, lny1b, lny2g, lny2b,
               Wly1, bly1, Wly2, bly2, Wly3, bly3,
               dis, h1, agg1, x1, h2, agg2, xln, t1, t2,
               Ab, Bb, Yfb, stream);

    // ---- final: out = Xf @ Yf^T, [10000,10000] fp32 --------------------------
    // 5 waves/block, 80x80 block tile, Y stripe staged in 20KB LDS.
    // Grid 125x125 (625 = 5*125 on both axes, exact tiling).
    gemm_final_wmma<5, 5, 128>
        <<<dim3(N / 80, N / 80), 5 * 32, 0, stream>>>(Xfb, Yfb, (float*)d_out, N);
}
